// LocalCorrPyramid_43198781063935
// MI455X (gfx1250) — matrix operand order repaired and checked by reading, not compiled, not run on hardware
//
#include <hip/hip_runtime.h>

#define H0 64
#define W0 96
#define D0 128
#define NLVL 4
#define KOFF 81
#define TPB 256   // 8 wave32 per workgroup

typedef _Float16 v16h __attribute__((ext_vector_type(16)));
typedef _Float16 v8h  __attribute__((ext_vector_type(8)));
typedef float    v8f  __attribute__((ext_vector_type(8)));

// ---------------------------------------------------------------------------
// Kernel 1: build one pyramid level. Exact f32 pooling from the ORIGINAL
// inputs (mean over 2^l x 2^l block), then convert features to f16,
// stored PIXEL-MAJOR [n][128] so WMMA A/B fragments are contiguous b128 loads.
// coords_l = mean(block)/2^l, kept f32, layout [2][n2].
// ---------------------------------------------------------------------------
__global__ void lcp_build_level(const float* __restrict__ f1,
                                const float* __restrict__ f2,
                                const float* __restrict__ cin,
                                _Float16* __restrict__ f1t,
                                _Float16* __restrict__ f2t,
                                float* __restrict__ cds,
                                int h, int w, int s) {
  int idx = blockIdx.x * blockDim.x + threadIdx.x;
  int n2 = h * w;
  if (idx >= n2 * D0) return;
  int n = idx >> 7;        // pixel
  int d = idx & 127;       // channel
  int y = n / w, x = n - y * w;
  const float inv = 1.0f / (float)(s * s);
  const int base = (y * s) * W0 + (x * s);
  const float* p1 = f1 + (size_t)d * (H0 * W0) + base;
  const float* p2 = f2 + (size_t)d * (H0 * W0) + base;
  float a1 = 0.0f, a2 = 0.0f;
  for (int i = 0; i < s; ++i)
    for (int j = 0; j < s; ++j) {
      a1 += p1[i * W0 + j];
      a2 += p2[i * W0 + j];
    }
  f1t[(size_t)n * D0 + d] = (_Float16)(a1 * inv);
  f2t[(size_t)n * D0 + d] = (_Float16)(a2 * inv);
  if (d < 2) {
    const float* pc = cin + (size_t)d * (H0 * W0) + base;
    float ac = 0.0f;
    for (int i = 0; i < s; ++i)
      for (int j = 0; j < s; ++j) ac += pc[i * W0 + j];
    cds[(size_t)d * n2 + n] = ac * inv / (float)s;  // pooled coords rescaled
  }
}

// ---------------------------------------------------------------------------
// Kernel 2: per-level correlation. One workgroup (8 wave32) owns 16 f1 pixels.
// Phase A: WMMA GEMM band[16][n2] = f1_tile(16x128) * f2^T(128xn2) / sqrt(128),
//          f32 accumulate, stored f16 in LDS (max 192 KB, fits CDNA5 320KB WGP LDS).
//          Column-tile loop is wave-uniform (readfirstlane) so EXEC is all-1s
//          at every v_wmma, per ISA requirement, and the loop is scalar-branched.
// Phase B: bilinear clamp-to-edge sampling of the LDS band for 16*81 outputs.
// ---------------------------------------------------------------------------
__global__ void __launch_bounds__(TPB) lcp_corr_level(
    const _Float16* __restrict__ f1t,
    const _Float16* __restrict__ f2t,
    const float* __restrict__ cds,
    float* __restrict__ out,
    int h, int w) {
  extern __shared__ _Float16 band[];  // [16][n2] f16
  const int n2   = h * w;
  const int tid  = threadIdx.x;
  const int lane = tid & 31;
  // wave id as an SGPR: guarantees the tile loop is uniform per wave.
  const int wave = __builtin_amdgcn_readfirstlane(tid >> 5);
  const int lrow = lane & 15;
  const int hsel = lane >> 4;     // 0: lanes 0-15, 1: lanes 16-31
  const int p0   = blockIdx.x * 16;

  // --- A fragments: 16 f1 rows x 128, split into 4 K-chunks of 32 ---
  // 16-bit A layout: lanes0-15 K = kc*32+{0..7,16..23}; lanes16-31 K = +8.
  const _Float16* arow = f1t + (size_t)(p0 + lrow) * D0;
  v16h afrag[4];
#pragma unroll
  for (int kc = 0; kc < 4; ++kc) {
    v8h alo = *(const v8h*)(arow + kc * 32 + hsel * 8);
    v8h ahi = *(const v8h*)(arow + kc * 32 + 16 + hsel * 8);
    afrag[kc] = __builtin_shufflevector(alo, ahi,
        0, 1, 2, 3, 4, 5, 6, 7, 8, 9, 10, 11, 12, 13, 14, 15);
  }

  const float scale = 0.08838834764831845f;  // 1/sqrt(128)

  // --- Phase A: band GEMM, 8 waves stride over 16-column tiles ---
  for (int n0 = wave * 16; n0 < n2; n0 += (TPB / 32) * 16) {
    // B layout: lanes0-15 col=n0+lrow K=kc*32+{0..7,16..23}; lanes16-31 K=+8.
    const _Float16* bcol = f2t + (size_t)(n0 + lrow) * D0 + hsel * 8;
    v8f acc = {0.f, 0.f, 0.f, 0.f, 0.f, 0.f, 0.f, 0.f};
#pragma unroll
    for (int kc = 0; kc < 4; ++kc) {
      v8h blo = *(const v8h*)(bcol + kc * 32);
      v8h bhi = *(const v8h*)(bcol + kc * 32 + 16);
      v16h bfrag = __builtin_shufflevector(blo, bhi,
          0, 1, 2, 3, 4, 5, 6, 7, 8, 9, 10, 11, 12, 13, 14, 15);
      acc = __builtin_amdgcn_wmma_f32_16x16x32_f16(
          false, afrag[kc], false, bfrag, (short)0, acc, false, false);
    }
    // D layout: VGPR r -> M = r + hsel*8, N = n0 + lrow
    const int ncol = n0 + lrow;
#pragma unroll
    for (int r = 0; r < 8; ++r) {
      band[(size_t)(r + hsel * 8) * n2 + ncol] = (_Float16)(acc[r] * scale);
    }
  }
  __syncthreads();

  // --- Phase B: bilinear sampling from LDS band (clamp-to-edge) ---
  for (int i = tid; i < 16 * KOFF; i += TPB) {
    int p = i / KOFF;
    int k = i - p * KOFF;
    int n = p0 + p;
    float cx = cds[n];
    float cy = cds[n2 + n];
    float xo = (float)(k / 9 - 4);   // meshgrid(indexing='ij'): xo slow, yo fast
    float yo = (float)(k % 9 - 4);
    float xs = fminf(fmaxf(cx + xo, 0.0f), (float)(w - 1));
    float ys = fminf(fmaxf(cy + yo, 0.0f), (float)(h - 1));
    float x0f = floorf(xs), y0f = floorf(ys);
    float wx = xs - x0f, wy = ys - y0f;
    int x0 = (int)x0f, y0 = (int)y0f;
    int x1 = (x0 + 1 < w - 1) ? (x0 + 1) : (w - 1);
    int y1 = (y0 + 1 < h - 1) ? (y0 + 1) : (h - 1);
    const _Float16* bp = band + (size_t)p * n2;
    float v00 = (float)bp[y0 * w + x0];
    float v01 = (float)bp[y0 * w + x1];
    float v10 = (float)bp[y1 * w + x0];
    float v11 = (float)bp[y1 * w + x1];
    float val = (1.0f - wx) * (1.0f - wy) * v00 + wx * (1.0f - wy) * v01 +
                (1.0f - wx) * wy * v10 + wx * wy * v11;
    out[(size_t)k * n2 + n] = val;
  }
}

// ---------------------------------------------------------------------------
extern "C" void kernel_launch(void* const* d_in, const int* in_sizes, int n_in,
                              void* d_out, int out_size, void* d_ws, size_t ws_size,
                              hipStream_t stream) {
  const float* f1 = (const float*)d_in[0];
  const float* f2 = (const float*)d_in[1];
  const float* co = (const float*)d_in[2];
  float* out = (float*)d_out;

  const int hh[NLVL]     = {64, 32, 16, 8};
  const int wwid[NLVL]   = {96, 48, 24, 12};
  const int pixoff[NLVL] = {0, 6144, 7680, 8064};  // prefix sums of n2
  const int totpix = 8160;

  // Workspace layout: f1t (f16 pixel-major), f2t (f16 pixel-major), coords (f32)
  _Float16* f1t = (_Float16*)d_ws;
  _Float16* f2t = f1t + (size_t)totpix * D0;
  float*    cds = (float*)(f2t + (size_t)totpix * D0);
  // total: 2*8160*128*2 + 8160*2*4 bytes ~= 4.25 MB

  for (int l = 0; l < NLVL; ++l) {
    int h = hh[l], w = wwid[l], n2 = h * w, s = 1 << l;
    int nthr = n2 * D0;
    int blocks = (nthr + 255) / 256;
    lcp_build_level<<<blocks, 256, 0, stream>>>(
        f1, f2, co,
        f1t + (size_t)pixoff[l] * D0,
        f2t + (size_t)pixoff[l] * D0,
        cds + (size_t)pixoff[l] * 2,
        h, w, s);
  }

  // Allow 192 KB dynamic LDS for the level-0 band (CDNA5 WGP has 320 KB).
  (void)hipFuncSetAttribute((const void*)lcp_corr_level,
                            hipFuncAttributeMaxDynamicSharedMemorySize,
                            16 * 6144 * (int)sizeof(_Float16));

  for (int l = 0; l < NLVL; ++l) {
    int h = hh[l], w = wwid[l], n2 = h * w;
    size_t shmem = (size_t)16 * n2 * sizeof(_Float16);
    lcp_corr_level<<<n2 / 16, TPB, shmem, stream>>>(
        f1t + (size_t)pixoff[l] * D0,
        f2t + (size_t)pixoff[l] * D0,
        cds + (size_t)pixoff[l] * 2,
        out + (size_t)KOFF * pixoff[l],
        h, w);
  }
}
